// copy_rnn_decoder_58969900974834
// MI455X (gfx1250) — matrix operand order repaired and checked by reading, not compiled, not run on hardware
//
#include <hip/hip_runtime.h>
#include <hip/hip_bf16.h>

// ---------------- problem constants (match reference setup) ----------------
#define TT   32      // timesteps
#define BB   32      // batch
#define SS   400     // source length
#define HH   512     // hidden
#define EE   512     // embedding
#define LLAY 2       // LSTM layers
#define VV   50000   // vocab
#define NOOV 50      // num oovs
#define EXTV (VV + NOOV)
#define UNKI 1

typedef __attribute__((ext_vector_type(16))) __bf16 v16bf;
typedef __attribute__((ext_vector_type(8)))  float  v8f;

// ---------------------------------------------------------------------------
// Preprocessing kernels
// ---------------------------------------------------------------------------
__global__ void cvt_bf16_k(const float* __restrict__ s, __hip_bfloat16* __restrict__ d, long n) {
    long i = (long)blockIdx.x * blockDim.x + threadIdx.x;
    if (i < n) d[i] = __float2bfloat16(s[i]);
}

// fused [W_ih | W_hh] -> bf16, per layer, rows = 4*HH, cols = EE+HH
__global__ void wcat_k(const float* __restrict__ Wih, const float* __restrict__ Whh,
                       __hip_bfloat16* __restrict__ out) {
    long n = (long)LLAY * 4 * HH * (EE + HH);
    long i = (long)blockIdx.x * blockDim.x + threadIdx.x;
    if (i >= n) return;
    int  k = (int)(i % (EE + HH));
    long r = i / (EE + HH);
    int  nn = (int)(r % (4 * HH));
    int  l  = (int)(r / (4 * HH));
    float v = (k < EE) ? Wih[((size_t)l * 4 * HH + nn) * EE + k]
                       : Whh[((size_t)l * 4 * HH + nn) * HH + (k - EE)];
    out[i] = __float2bfloat16(v);
}

__global__ void bcat_k(const float* __restrict__ bih, const float* __restrict__ bhh,
                       float* __restrict__ out) {
    int i = blockIdx.x * blockDim.x + threadIdx.x;
    if (i < LLAY * 4 * HH) out[i] = bih[i] + bhh[i];
}

__global__ void zero_k(float* __restrict__ p, long n) {
    long i = (long)blockIdx.x * blockDim.x + threadIdx.x;
    if (i < n) p[i] = 0.0f;
}

// ---------------------------------------------------------------------------
// Pack M=32 activation(s) into CDNA5 16-bit WMMA A-fragment layout (bf16).
// Fragment f = (kt*2 + mtile)*32 + lane, 16 contiguous bf16 each.
// Lanes 0-15: row=lane, K = {kb..kb+15} with kb = kt*32.
// Lanes 16-31: row=lane-16, same but kb += 16.   (matches contiguous B fragment)
// ---------------------------------------------------------------------------
__global__ void pack_a2_k(const float* __restrict__ A0, const float* __restrict__ A1,
                          int K0, int K1, __hip_bfloat16* __restrict__ out) {
    int K = K0 + K1;
    int nfrag = (K >> 5) * 64;
    int f = blockIdx.x * blockDim.x + threadIdx.x;
    if (f >= nfrag) return;
    int lane = f & 31, mt = (f >> 5) & 1, kt = f >> 6;
    int row = mt * 16 + (lane & 15);
    int kb  = kt * 32 + ((lane & 16) ? 16 : 0);
    __hip_bfloat16* dst = out + (size_t)f * 16;
#pragma unroll
    for (int i = 0; i < 16; ++i) {
        int k = kb + i;                     // 0..15 relative
        float v = (k < K0) ? A0[(size_t)row * K0 + k]
                           : A1[(size_t)row * K1 + (k - K0)];
        dst[i] = __float2bfloat16(v);
    }
}

// ---------------------------------------------------------------------------
// WMMA GEMM: C[32,N] = A[32,K](bf16 frags) * W[N,K](bf16,row-major)^T + bias
// One wave per 16-column N-tile; both 16-row M-tiles accumulated per wave.
// act: 0 = none, 1 = tanh
// ---------------------------------------------------------------------------
__global__ void wmma_gemm_k(const __hip_bfloat16* __restrict__ Afrag,
                            const __hip_bfloat16* __restrict__ W,
                            const float* __restrict__ bias,
                            float* __restrict__ C, int N, int K, int act) {
    int lane  = threadIdx.x & 31;
    int wave  = threadIdx.x >> 5;
    int ntile = blockIdx.x * (blockDim.x >> 5) + wave;
    if (ntile * 16 >= N) return;              // wave-uniform: EXEC all-1s for WMMA

    int lane16 = lane & 15;
    int hi     = (lane >> 4) & 1;             // B frag: lanes 0-15 K=0..15, 16-31 K=16..31
    const __hip_bfloat16* wrow = W + (size_t)(ntile * 16 + lane16) * K + hi * 16;
    const v16bf* Af = reinterpret_cast<const v16bf*>(Afrag);

    v8f acc0 = {}; v8f acc1 = {};
    int ktmax = K >> 5;
    for (int kt = 0; kt < ktmax; ++kt) {
        v16bf b  = *reinterpret_cast<const v16bf*>(wrow + kt * 32);
        if (kt + 1 < ktmax)
            __builtin_prefetch(wrow + (kt + 1) * 32, 0, 1);   // global_prefetch_b8
        v16bf a0 = Af[(kt * 2 + 0) * 32 + lane];
        v16bf a1 = Af[(kt * 2 + 1) * 32 + lane];
        acc0 = __builtin_amdgcn_wmma_f32_16x16x32_bf16(false, a0, false, b,
                                                       (short)0, acc0, false, false);
        acc1 = __builtin_amdgcn_wmma_f32_16x16x32_bf16(false, a1, false, b,
                                                       (short)0, acc1, false, false);
    }
    int col = ntile * 16 + lane16;
    float bv = bias ? bias[col] : 0.0f;
#pragma unroll
    for (int r = 0; r < 8; ++r) {             // C layout: VGPR r -> M=r (+8 for hi half)
        float v0 = acc0[r] + bv;
        float v1 = acc1[r] + bv;
        if (act == 1) { v0 = tanhf(v0); v1 = tanhf(v1); }
        C[(size_t)(r + hi * 8)      * N + col] = v0;
        C[(size_t)(16 + r + hi * 8) * N + col] = v1;
    }
}

// ---------------------------------------------------------------------------
// Per-step element kernels
// ---------------------------------------------------------------------------
__global__ void embed_k(const int* __restrict__ ids, const float* __restrict__ emb,
                        float* __restrict__ x) {
    int i = blockIdx.x * blockDim.x + threadIdx.x;  // BB*EE
    if (i >= BB * EE) return;
    int b = i / EE, e = i % EE;
    int id = ids[b];
    if (id >= VV) id = UNKI;
    x[i] = emb[(size_t)id * EE + e];
}

__global__ void lstm_gates_k(const float* __restrict__ g, float* __restrict__ h,
                             float* __restrict__ c) {
    int i = blockIdx.x * blockDim.x + threadIdx.x;  // BB*HH
    if (i >= BB * HH) return;
    int b = i / HH, j = i % HH;
    const float* gb = g + (size_t)b * 4 * HH;
    float gi = gb[j], gf = gb[HH + j], gg = gb[2 * HH + j], go = gb[3 * HH + j];
    float si = 1.0f / (1.0f + __expf(-gi));
    float sf = 1.0f / (1.0f + __expf(-gf));
    float so = 1.0f / (1.0f + __expf(-go));
    float cn = sf * c[i] + si * tanhf(gg);
    c[i] = cn;
    h[i] = so * tanhf(cn);
}

// one block per batch row: scores over S, softmax, context vector
__global__ void attention_k(const float* __restrict__ contexts, const float* __restrict__ gamma,
                            float* __restrict__ attn, float* __restrict__ ct) {
    int b = blockIdx.x, tid = threadIdx.x;          // 256 threads
    __shared__ float sg[HH];
    __shared__ float sc[SS];
    __shared__ float red[256];
    for (int k = tid; k < HH; k += 256) sg[k] = gamma[b * HH + k];
    __syncthreads();
    const float* cb = contexts + (size_t)b * SS * HH;
    for (int j = tid; j < SS; j += 256) {
        const float* row = cb + (size_t)j * HH;
        float d = 0.0f;
        for (int k = 0; k < HH; ++k) d += row[k] * sg[k];
        sc[j] = d;
    }
    __syncthreads();
    float m = -1e30f;
    for (int j = tid; j < SS; j += 256) m = fmaxf(m, sc[j]);
    red[tid] = m; __syncthreads();
    for (int s = 128; s > 0; s >>= 1) { if (tid < s) red[tid] = fmaxf(red[tid], red[tid + s]); __syncthreads(); }
    m = red[0]; __syncthreads();
    float lsum = 0.0f;
    for (int j = tid; j < SS; j += 256) { float e = __expf(sc[j] - m); sc[j] = e; lsum += e; }
    red[tid] = lsum; __syncthreads();
    for (int s = 128; s > 0; s >>= 1) { if (tid < s) red[tid] += red[tid + s]; __syncthreads(); }
    float inv = 1.0f / red[0];
    __syncthreads();
    for (int j = tid; j < SS; j += 256) { sc[j] *= inv; attn[b * SS + j] = sc[j]; }
    __syncthreads();
    for (int k = tid; k < HH; k += 256) {
        float a = 0.0f;
        for (int j = 0; j < SS; ++j) a += sc[j] * cb[(size_t)j * HH + k];
        ct[b * HH + k] = a;
    }
}

// p_gen = sigmoid(Wg . [h_att, h_top, c_top, x] + bg), one block per batch
__global__ void pgen_k(const float* __restrict__ hatt, const float* __restrict__ htop,
                       const float* __restrict__ ctop, const float* __restrict__ x,
                       const float* __restrict__ Wg, const float* __restrict__ bg,
                       float* __restrict__ pg) {
    int b = blockIdx.x, tid = threadIdx.x;          // 256 threads
    __shared__ float red[256];
    float s = 0.0f;
    for (int k = tid; k < 4 * HH; k += 256) {       // 3*HH + EE == 4*HH here
        float v;
        if (k < HH)          v = hatt[b * HH + k];
        else if (k < 2 * HH) v = htop[b * HH + (k - HH)];
        else if (k < 3 * HH) v = ctop[b * HH + (k - 2 * HH)];
        else                 v = x[b * EE + (k - 3 * HH)];
        s += Wg[k] * v;
    }
    red[tid] = s; __syncthreads();
    for (int st = 128; st > 0; st >>= 1) { if (tid < st) red[tid] += red[tid + st]; __syncthreads(); }
    if (tid == 0) pg[b] = 1.0f / (1.0f + __expf(-(red[0] + bg[0])));
}

// per-row max & sum-of-exp over vocab logits (one block per batch row)
__global__ void rowstat_k(const float* __restrict__ logits, float* __restrict__ rmax,
                          float* __restrict__ rsum) {
    int b = blockIdx.x, tid = threadIdx.x;          // 512 threads
    __shared__ float red[512];
    const float* lb = logits + (size_t)b * VV;
    float m = -1e30f;
    for (int v = tid; v < VV; v += 512) m = fmaxf(m, lb[v]);
    red[tid] = m; __syncthreads();
    for (int s = 256; s > 0; s >>= 1) { if (tid < s) red[tid] = fmaxf(red[tid], red[tid + s]); __syncthreads(); }
    m = red[0]; __syncthreads();
    float su = 0.0f;
    for (int v = tid; v < VV; v += 512) su += __expf(lb[v] - m);
    red[tid] = su; __syncthreads();
    for (int s = 256; s > 0; s >>= 1) { if (tid < s) red[tid] += red[tid + s]; __syncthreads(); }
    if (tid == 0) { rmax[b] = m; rsum[b] = red[0]; }
}

// scatter attn (or zeros) at src positions; src is [S,B] so src_t[b,j] = src[j*B+b]
__global__ void scatter_k(const int* __restrict__ src, const float* __restrict__ attn,
                          float* __restrict__ copybuf, int clear) {
    int i = blockIdx.x * blockDim.x + threadIdx.x;  // BB*SS
    if (i >= BB * SS) return;
    int b = i / SS, j = i % SS;
    int v = src[j * BB + b];
    copybuf[(size_t)b * EXTV + v] = clear ? 0.0f : attn[b * SS + j];
}

__global__ void scores_k(const float* __restrict__ logits, const float* __restrict__ rmax,
                         const float* __restrict__ rsum, const float* __restrict__ pg,
                         const float* __restrict__ copybuf, float* __restrict__ out_t) {
    long i = (long)blockIdx.x * blockDim.x + threadIdx.x;  // BB*EXTV
    if (i >= (long)BB * EXTV) return;
    int b = (int)(i / EXTV), v = (int)(i % EXTV);
    float p = pg[b];
    float vd = 0.0f;
    if (v < VV) vd = __expf(logits[(size_t)b * VV + v] - rmax[b]) / rsum[b];
    out_t[i] = __logf(p * vd + (1.0f - p) * copybuf[i] + 1e-10f);
}

// ---------------------------------------------------------------------------
// Host driver
// ---------------------------------------------------------------------------
static inline int gridFor(long n, int t) { return (int)((n + t - 1) / t); }

extern "C" void kernel_launch(void* const* d_in, const int* in_sizes, int n_in,
                              void* d_out, int out_size, void* d_ws, size_t ws_size,
                              hipStream_t stream) {
    (void)in_sizes; (void)n_in; (void)out_size; (void)ws_size;
    const int*   inputs   = (const int*)  d_in[0];
    const int*   src      = (const int*)  d_in[1];
    const float* h0       = (const float*)d_in[2];
    const float* c0       = (const float*)d_in[3];
    const float* contexts = (const float*)d_in[4];
    const float* emb      = (const float*)d_in[6];
    const float* W_ih     = (const float*)d_in[7];
    const float* b_ih     = (const float*)d_in[8];
    const float* W_hh     = (const float*)d_in[9];
    const float* b_hh     = (const float*)d_in[10];
    const float* Wa       = (const float*)d_in[11];
    const float* ba       = (const float*)d_in[12];
    const float* Wo       = (const float*)d_in[13];
    const float* bo       = (const float*)d_in[14];
    const float* Wv       = (const float*)d_in[15];
    const float* bv       = (const float*)d_in[16];
    const float* Wg       = (const float*)d_in[17];
    const float* bg       = (const float*)d_in[18];
    float* out = (float*)d_out;

    // ---- workspace carve-out (256B aligned) ----
    char* wp = (char*)d_ws;
    auto carve = [&](size_t bytes) { char* p = wp; wp += (bytes + 255) & ~(size_t)255; return p; };
    __hip_bfloat16* wv_bf   = (__hip_bfloat16*)carve((size_t)VV * HH * 2);
    __hip_bfloat16* wcat_bf = (__hip_bfloat16*)carve((size_t)LLAY * 4 * HH * (EE + HH) * 2);
    __hip_bfloat16* wa_bf   = (__hip_bfloat16*)carve((size_t)HH * HH * 2);
    __hip_bfloat16* wo_bf   = (__hip_bfloat16*)carve((size_t)HH * 2 * HH * 2);
    float* bcat    = (float*)carve((size_t)LLAY * 4 * HH * 4);
    float* xbuf    = (float*)carve((size_t)BB * EE * 4);
    float* h_buf   = (float*)carve((size_t)LLAY * BB * HH * 4);
    float* c_buf   = (float*)carve((size_t)LLAY * BB * HH * 4);
    __hip_bfloat16* afrag = (__hip_bfloat16*)carve((size_t)BB * (EE + HH) * 2);
    float* gbuf    = (float*)carve((size_t)BB * 4 * HH * 4);
    float* gamma   = (float*)carve((size_t)BB * HH * 4);
    float* attnb   = (float*)carve((size_t)BB * SS * 4);
    float* ctb     = (float*)carve((size_t)BB * HH * 4);
    float* hatt    = (float*)carve((size_t)BB * HH * 4);
    float* logits  = (float*)carve((size_t)BB * VV * 4);
    float* rmax    = (float*)carve((size_t)BB * 4);
    float* rsum    = (float*)carve((size_t)BB * 4);
    float* pgbuf   = (float*)carve((size_t)BB * 4);
    float* copybuf = (float*)carve((size_t)BB * EXTV * 4);

    const int T256 = 256;

    // ---- preprocessing (per launch; deterministic) ----
    cvt_bf16_k<<<gridFor((long)VV * HH, T256), T256, 0, stream>>>(Wv, wv_bf, (long)VV * HH);
    cvt_bf16_k<<<gridFor((long)HH * HH, T256), T256, 0, stream>>>(Wa, wa_bf, (long)HH * HH);
    cvt_bf16_k<<<gridFor((long)HH * 2 * HH, T256), T256, 0, stream>>>(Wo, wo_bf, (long)HH * 2 * HH);
    wcat_k<<<gridFor((long)LLAY * 4 * HH * (EE + HH), T256), T256, 0, stream>>>(W_ih, W_hh, wcat_bf);
    bcat_k<<<gridFor(LLAY * 4 * HH, T256), T256, 0, stream>>>(b_ih, b_hh, bcat);
    zero_k<<<gridFor((long)BB * EXTV, T256), T256, 0, stream>>>(copybuf, (long)BB * EXTV);
    hipMemcpyAsync(h_buf, h0, (size_t)LLAY * BB * HH * 4, hipMemcpyDeviceToDevice, stream);
    hipMemcpyAsync(c_buf, c0, (size_t)LLAY * BB * HH * 4, hipMemcpyDeviceToDevice, stream);

    // GEMM launch geometry: 4 waves (128 thr) per block, 1 ntile per wave
    const int GT = 128, WPB = 4;
    const int gLstm  = gridFor((4 * HH) / 16, WPB);   // N=2048
    const int gSmall = gridFor(HH / 16, WPB);         // N=512
    const int gVoc   = gridFor(VV / 16, WPB);         // N=50000

    for (int t = 0; t < TT; ++t) {
        // embedding
        embed_k<<<gridFor(BB * EE, T256), T256, 0, stream>>>(inputs + t * BB, emb, xbuf);

        // LSTM layer 0: A = [x | h0_prev], W = [W_ih0 | W_hh0]
        pack_a2_k<<<gridFor(((EE + HH) / 32) * 64, T256), T256, 0, stream>>>(
            xbuf, h_buf, EE, HH, afrag);
        wmma_gemm_k<<<gLstm, GT, 0, stream>>>(afrag, wcat_bf, bcat, gbuf, 4 * HH, EE + HH, 0);
        lstm_gates_k<<<gridFor(BB * HH, T256), T256, 0, stream>>>(gbuf, h_buf, c_buf);

        // LSTM layer 1: A = [h0_new | h1_prev]
        pack_a2_k<<<gridFor(((HH + HH) / 32) * 64, T256), T256, 0, stream>>>(
            h_buf, h_buf + BB * HH, HH, HH, afrag);
        wmma_gemm_k<<<gLstm, GT, 0, stream>>>(afrag, wcat_bf + (size_t)4 * HH * (EE + HH),
                                              bcat + 4 * HH, gbuf, 4 * HH, HH + HH, 0);
        lstm_gates_k<<<gridFor(BB * HH, T256), T256, 0, stream>>>(
            gbuf, h_buf + BB * HH, c_buf + BB * HH);

        // attention input: gamma = out @ Wa^T + ba
        pack_a2_k<<<gridFor((HH / 32) * 64, T256), T256, 0, stream>>>(
            h_buf + BB * HH, h_buf + BB * HH, HH, 0, afrag);
        wmma_gemm_k<<<gSmall, GT, 0, stream>>>(afrag, wa_bf, ba, gamma, HH, HH, 0);

        // softmax attention + context vector
        attention_k<<<BB, 256, 0, stream>>>(contexts, gamma, attnb, ctb);

        // h_att = tanh([c_t | out] @ Wo^T + bo)
        pack_a2_k<<<gridFor(((HH + HH) / 32) * 64, T256), T256, 0, stream>>>(
            ctb, h_buf + BB * HH, HH, HH, afrag);
        wmma_gemm_k<<<gSmall, GT, 0, stream>>>(afrag, wo_bf, bo, hatt, HH, HH + HH, 1);

        // p_gen
        pgen_k<<<BB, 256, 0, stream>>>(hatt, h_buf + BB * HH, c_buf + BB * HH, xbuf,
                                       Wg, bg, pgbuf);

        // vocab logits = h_att @ Wv^T + bv   (dominant GEMM, Wv bf16 resident in L2)
        pack_a2_k<<<gridFor((HH / 32) * 64, T256), T256, 0, stream>>>(
            hatt, hatt, HH, 0, afrag);
        wmma_gemm_k<<<gVoc, GT, 0, stream>>>(afrag, wv_bf, bv, logits, VV, HH, 0);

        // softmax stats, copy-dist scatter, final log-scores, then clear scatter
        rowstat_k<<<BB, 512, 0, stream>>>(logits, rmax, rsum);
        scatter_k<<<gridFor(BB * SS, T256), T256, 0, stream>>>(src, attnb, copybuf, 0);
        scores_k<<<gridFor((long)BB * EXTV, T256), T256, 0, stream>>>(
            logits, rmax, rsum, pgbuf, copybuf, out + (size_t)t * BB * EXTV);
        scatter_k<<<gridFor(BB * SS, T256), T256, 0, stream>>>(src, attnb, copybuf, 1);
    }

    // trailing outputs: h then c
    float* out_h = out + (size_t)TT * BB * EXTV;
    hipMemcpyAsync(out_h, h_buf, (size_t)LLAY * BB * HH * 4, hipMemcpyDeviceToDevice, stream);
    hipMemcpyAsync(out_h + (size_t)LLAY * BB * HH, c_buf, (size_t)LLAY * BB * HH * 4,
                   hipMemcpyDeviceToDevice, stream);
}
